// Attention_84250078478891
// MI455X (gfx1250) — matrix operand order repaired and checked by reading, not compiled, not run on hardware
//
#include <hip/hip_runtime.h>
#include <hip/hip_bf16.h>
#include <stddef.h>
#include <stdint.h>

// ---------------- problem constants ----------------
constexpr int Bc   = 2;
constexpr int Sc   = 2048;
constexpr int Hc   = 4096;
constexpr int NHc  = 32;
constexpr int NKVc = 2;
constexpr int HDc  = 128;
constexpr int ROTc = 64;
constexpr int Gc   = NHc / NKVc;             // 16
constexpr int Nqkv = (NHc + 2 * NKVc) * HDc; // 4608
constexpr int Mrows = Bc * Sc;               // 4096

typedef __attribute__((ext_vector_type(16))) __bf16 v16bf;
typedef __attribute__((ext_vector_type(8)))  __bf16 v8bf;
typedef __attribute__((ext_vector_type(8)))  float  v8f;

// ---------------- helpers ----------------
__device__ __host__ inline __bf16 f2bf(float f) {
  union { float f; unsigned int u; } v; v.f = f;
  unsigned int r = (v.u + 0x7FFFu + ((v.u >> 16) & 1u)) >> 16;
  union { unsigned short u; __bf16 b; } o; o.u = (unsigned short)r;
  return o.b;
}
__device__ inline float bf2f(__bf16 x) {
  union { unsigned short u; __bf16 b; } t; t.b = x;
  union { unsigned int u; float f; } o; o.u = ((unsigned int)t.u) << 16;
  return o.f;
}
__device__ inline v16bf cat8(v8bf lo, v8bf hi) {
  v16bf r;
#pragma unroll
  for (int i = 0; i < 8; i++) { r[i] = lo[i]; r[i + 8] = hi[i]; }
  return r;
}
__device__ inline v8f wmma_bf16(v16bf a, v16bf b, v8f c) {
  return __builtin_amdgcn_wmma_f32_16x16x32_bf16(
      false, a, false, b, (short)0, c, false, false);
}

// CDNA5 async copy: global -> LDS, 16B per lane, tracked by ASYNCcnt.
// VDST holds the wave-relative LDS byte offset (low 32 bits of the flat
// shared-aperture address), VADDR holds the 64-bit global address.
__device__ inline void async_b128(const void* gaddr, void* ldsdst) {
  unsigned loff = (unsigned)(uintptr_t)ldsdst;
  unsigned long long ga = (unsigned long long)(uintptr_t)gaddr;
  asm volatile("global_load_async_to_lds_b128 %0, %1, off"
               :: "v"(loff), "v"(ga) : "memory");
}
__device__ inline void wait_async0() {
  asm volatile("s_wait_asynccnt 0" ::: "memory");
}

// ---------------- fp32 -> bf16 convert ----------------
__global__ void cvt_f32_bf16(const float* __restrict__ src,
                             __bf16* __restrict__ dst, size_t n) {
  size_t i = (size_t)blockIdx.x * blockDim.x + threadIdx.x;
  size_t stride = (size_t)gridDim.x * blockDim.x;
  for (; i < n; i += stride) dst[i] = f2bf(src[i]);
}

// ---------------- tiled bf16 WMMA GEMM ----------------
// C[M,N] = A[M,K] * B[K,N] (+bias). 256 threads, 128x128 tile, BK=32.
// Double-buffered LDS; A staged via async-to-LDS DMA, B register-pipelined.
template <bool HAS_BIAS, bool OUT_BF16>
__global__ __launch_bounds__(256) void gemm_wmma(
    const __bf16* __restrict__ A, const __bf16* __restrict__ Bw,
    const float* __restrict__ bias, void* __restrict__ Cout,
    int M, int N, int K) {
  constexpr int LDA = 40, LDB = 40;
  __shared__ __align__(16) __bf16 Asm[2][128 * LDA];
  __shared__ __align__(16) __bf16 Bsm[2][128 * LDB];

  const int tid = threadIdx.x;
  const int lane = tid & 31, wid = tid >> 5;
  const int lane16 = lane & 15, half = lane >> 4;
  const int wm = wid & 1, wn = wid >> 1;
  const int m0 = blockIdx.y * 128, n0 = blockIdx.x * 128;

  const int am = tid >> 1, akk = (tid & 1) * 16;  // A staging: half-row of 16
  const int bk = tid >> 3, bnn = (tid & 7) * 16;  // B staging: 16-elem chunk

  v8f acc[4][2];
  const v8f vz = {0.f, 0.f, 0.f, 0.f, 0.f, 0.f, 0.f, 0.f};
#pragma unroll
  for (int i = 0; i < 4; i++)
#pragma unroll
    for (int j = 0; j < 2; j++) acc[i][j] = vz;

  const int nk = K / 32;

  // ---- prologue: tile 0 ----
  v8bf br0, br1;
  {
    const __bf16* bsrc = Bw + (size_t)bk * N + n0 + bnn;
    br0 = *(const v8bf*)(bsrc);
    br1 = *(const v8bf*)(bsrc + 8);
    const __bf16* asrc = A + (size_t)(m0 + am) * K + akk;
    async_b128(asrc, &Asm[0][am * LDA + akk]);
    async_b128(asrc + 8, &Asm[0][am * LDA + akk + 8]);
  }

  for (int kt = 0; kt < nk; kt++) {
    const int cur = kt & 1;
    // scatter B registers (transposed) into current buffer
#pragma unroll
    for (int e = 0; e < 8; e++) {
      Bsm[cur][(bnn + e) * LDB + bk] = br0[e];
      Bsm[cur][(bnn + 8 + e) * LDB + bk] = br1[e];
    }
    wait_async0();        // A tile for `cur` has landed in LDS
    __syncthreads();      // publish A + B staging to all waves

    // prefetch tile kt+1 (async DMA + B regs) overlapping the WMMAs below
    if (kt + 1 < nk) {
      const int k0n = (kt + 1) * 32;
      const __bf16* bsrc = Bw + (size_t)(k0n + bk) * N + n0 + bnn;
      br0 = *(const v8bf*)(bsrc);
      br1 = *(const v8bf*)(bsrc + 8);
      const __bf16* asrc = A + (size_t)(m0 + am) * K + k0n + akk;
      async_b128(asrc, &Asm[1 - cur][am * LDA + akk]);
      async_b128(asrc + 8, &Asm[1 - cur][am * LDA + akk + 8]);
    }

    // batch all fragment loads, then issue the 8 WMMAs
    v16bf af[4], bf[2];
#pragma unroll
    for (int mt = 0; mt < 4; mt++) {
      const int ml = wm * 64 + mt * 16 + lane16;
      v8bf lo = *(const v8bf*)&Asm[cur][ml * LDA + 8 * half];
      v8bf hi = *(const v8bf*)&Asm[cur][ml * LDA + 16 + 8 * half];
      af[mt] = cat8(lo, hi);
    }
#pragma unroll
    for (int nt = 0; nt < 2; nt++) {
      const int nl = wn * 32 + nt * 16 + lane16;
      v8bf lo = *(const v8bf*)&Bsm[cur][nl * LDB + 8 * half];
      v8bf hi = *(const v8bf*)&Bsm[cur][nl * LDB + 16 + 8 * half];
      bf[nt] = cat8(lo, hi);
    }
#pragma unroll
    for (int mt = 0; mt < 4; mt++)
#pragma unroll
      for (int nt = 0; nt < 2; nt++)
        acc[mt][nt] = wmma_bf16(af[mt], bf[nt], acc[mt][nt]);
  }

  // epilogue
#pragma unroll
  for (int mt = 0; mt < 4; mt++) {
#pragma unroll
    for (int nt = 0; nt < 2; nt++) {
#pragma unroll
      for (int r = 0; r < 8; r++) {
        const int row = m0 + wm * 64 + mt * 16 + r + 8 * half;
        const int col = n0 + wn * 32 + nt * 16 + lane16;
        float v = acc[mt][nt][r];
        if (HAS_BIAS) v += bias[col];
        if (OUT_BF16)
          ((__bf16*)Cout)[(size_t)row * N + col] = f2bf(v);
        else
          ((float*)Cout)[(size_t)row * N + col] = v;
      }
    }
  }
}

// ---------------- RoPE + layout split ----------------
__global__ __launch_bounds__(128) void rope_split(
    const __bf16* __restrict__ qkvb, const int* __restrict__ positions,
    __bf16* __restrict__ Qb, __bf16* __restrict__ Kb, __bf16* __restrict__ Vb) {
  const int bs = blockIdx.x;
  const int b = bs / Sc, s = bs % Sc;
  const float pos = (float)positions[bs];
  const __bf16* row = qkvb + (size_t)bs * Nqkv;
  constexpr float LOG1E4 = 9.210340371976184f; // ln(10000)

  for (int idx = threadIdx.x; idx < NHc * HDc / 2; idx += 128) {
    const int h = idx >> 6, p = idx & 63;
    const int d0 = 2 * p;
    float x1 = bf2f(row[h * HDc + d0]);
    float x2 = bf2f(row[h * HDc + d0 + 1]);
    float o1 = x1, o2 = x2;
    if (d0 < ROTc) {
      float inv = __expf(-LOG1E4 * (float)p * (1.0f / 32.0f));
      float ang = pos * inv;
      float cc = __cosf(ang), ss = __sinf(ang);
      o1 = x1 * cc - x2 * ss;
      o2 = x1 * ss + x2 * cc;
    }
    __bf16* dst = Qb + ((size_t)(b * NHc + h) * Sc + s) * HDc + d0;
    dst[0] = f2bf(o1);
    dst[1] = f2bf(o2);
  }
  for (int idx = threadIdx.x; idx < NKVc * HDc / 2; idx += 128) {
    const int g = idx >> 6, p = idx & 63;
    const int d0 = 2 * p;
    const size_t src = (size_t)NHc * HDc + g * HDc + d0;
    float x1 = bf2f(row[src]);
    float x2 = bf2f(row[src + 1]);
    float o1 = x1, o2 = x2;
    if (d0 < ROTc) {
      float inv = __expf(-LOG1E4 * (float)p * (1.0f / 32.0f));
      float ang = pos * inv;
      float cc = __cosf(ang), ss = __sinf(ang);
      o1 = x1 * cc - x2 * ss;
      o2 = x1 * ss + x2 * cc;
    }
    __bf16* dst = Kb + ((size_t)(b * NKVc + g) * Sc + s) * HDc + d0;
    dst[0] = f2bf(o1);
    dst[1] = f2bf(o2);
  }
  for (int idx = threadIdx.x; idx < NKVc * HDc; idx += 128) {
    const int g = idx / HDc, d = idx % HDc;
    Vb[((size_t)(b * NKVc + g) * Sc + s) * HDc + d] =
        row[(size_t)(NHc + NKVc) * HDc + g * HDc + d];
  }
}

// ---------------- flash attention (causal, GQA) ----------------
// grid: (S/64, NH, B), 128 threads (4 waves). Wave w owns 16 query rows.
// K tiles: async-to-LDS DMA, double-buffered, shared by all 4 waves.
// V tiles: register-pipelined transposed staging, double-buffered.
__global__ __launch_bounds__(128) void attn_flash(
    const __bf16* __restrict__ Q, const __bf16* __restrict__ Kt,
    const __bf16* __restrict__ Vt, __bf16* __restrict__ ctx) {
  constexpr int LDK = 136; // 32x128 K tile rows, padded (16B-aligned stride)
  constexpr int LDV = 40;  // 128x32 V^T tile rows, padded
  __shared__ __align__(16) __bf16 Ksm[2][32 * LDK];
  __shared__ __align__(16) __bf16 VTs[2][HDc * LDV];
  __shared__ __align__(16) __bf16 Psm[4][16 * 32];

  const int tid = threadIdx.x;
  const int wid = tid >> 5, lane = tid & 31;
  const int lane16 = lane & 15, half = lane >> 4;
  const int qb = blockIdx.x, head = blockIdx.y, b = blockIdx.z;
  const int g = head / Gc;
  const int qrow0 = qb * 64 + wid * 16;
  const float scale = 0.08838834764831845f; // 1/sqrt(128)
  const float MASKV = -3.0e38f;

  // staging responsibilities: 128 threads cover a 32x128 tile, 32 elems each
  const int skey = tid >> 2;       // key row 0..31
  const int shd = (tid & 3) * 32;  // 32-elem segment of that row

  const __bf16* kbase = Kt + (size_t)(b * NKVc + g) * Sc * HDc;
  const __bf16* vbase = Vt + (size_t)(b * NKVc + g) * Sc * HDc;

  // Q A-fragments
  const __bf16* qptr =
      Q + ((size_t)(b * NHc + head) * Sc + qrow0 + lane16) * HDc;
  v16bf qf[4];
#pragma unroll
  for (int t = 0; t < 4; t++) {
    v8bf lo = *(const v8bf*)(qptr + 32 * t + 8 * half);
    v8bf hi = *(const v8bf*)(qptr + 32 * t + 16 + 8 * half);
    qf[t] = cat8(lo, hi);
  }

  const v8f vz = {0.f, 0.f, 0.f, 0.f, 0.f, 0.f, 0.f, 0.f};
  v8f O[8];
#pragma unroll
  for (int nt = 0; nt < 8; nt++) O[nt] = vz;
  float mrow[8], lrow[8];
#pragma unroll
  for (int r = 0; r < 8; r++) { mrow[r] = MASKV; lrow[r] = 0.f; }

  const int nj = (qb + 1) * 2; // 32-key blocks up to end of query block

  // ---- prologue: tile 0 ----
  v8bf vreg[4];
  {
    const __bf16* vsrc = vbase + (size_t)skey * HDc + shd;
#pragma unroll
    for (int i = 0; i < 4; i++) vreg[i] = *(const v8bf*)(vsrc + 8 * i);
    const __bf16* ksrc = kbase + (size_t)skey * HDc + shd;
#pragma unroll
    for (int i = 0; i < 4; i++)
      async_b128(ksrc + 8 * i, &Ksm[0][skey * LDK + shd + 8 * i]);
  }

  for (int jb = 0; jb < nj; jb++) {
    const int cur = jb & 1;
    // scatter V registers transposed into current buffer
#pragma unroll
    for (int e = 0; e < 32; e++)
      VTs[cur][(shd + e) * LDV + skey] = vreg[e >> 3][e & 7];
    wait_async0();        // K tile for `cur` has landed in LDS
    __syncthreads();      // publish K + V^T staging

    // prefetch tile jb+1 while WMMAs below execute
    if (jb + 1 < nj) {
      const int j0n = (jb + 1) * 32;
      const __bf16* vsrc = vbase + (size_t)(j0n + skey) * HDc + shd;
#pragma unroll
      for (int i = 0; i < 4; i++) vreg[i] = *(const v8bf*)(vsrc + 8 * i);
      const __bf16* ksrc = kbase + (size_t)(j0n + skey) * HDc + shd;
#pragma unroll
      for (int i = 0; i < 4; i++)
        async_b128(ksrc + 8 * i, &Ksm[1 - cur][skey * LDK + shd + 8 * i]);
    }

    const int j0 = jb * 32;
    // ---- scores: batch all 8 K-fragment loads, then 8 WMMAs ----
    v16bf kfr[2][4];
#pragma unroll
    for (int kt = 0; kt < 2; kt++) {
      const __bf16* krow = &Ksm[cur][(kt * 16 + lane16) * LDK];
#pragma unroll
      for (int t = 0; t < 4; t++) {
        v8bf lo = *(const v8bf*)&krow[32 * t + 8 * half];
        v8bf hi = *(const v8bf*)&krow[32 * t + 16 + 8 * half];
        kfr[kt][t] = cat8(lo, hi);
      }
    }
    v8f s0 = vz, s1 = vz;
#pragma unroll
    for (int t = 0; t < 4; t++) s0 = wmma_bf16(qf[t], kfr[0][t], s0);
#pragma unroll
    for (int t = 0; t < 4; t++) s1 = wmma_bf16(qf[t], kfr[1][t], s1);

    // ---- online softmax over this 32-key block ----
#pragma unroll
    for (int r = 0; r < 8; r++) {
      const int qrow = qrow0 + r + 8 * half;
      const int k0i = j0 + lane16, k1i = j0 + 16 + lane16;
      float a0 = s0[r] * scale; if (k0i > qrow) a0 = MASKV;
      float a1 = s1[r] * scale; if (k1i > qrow) a1 = MASKV;
      float vmax = fmaxf(a0, a1);
#pragma unroll
      for (int off = 1; off < 16; off <<= 1)
        vmax = fmaxf(vmax, __shfl_xor(vmax, off, 32));
      const float mnew = fmaxf(mrow[r], vmax);
      const float p0 = __expf(a0 - mnew);
      const float p1 = __expf(a1 - mnew);
      float psum = p0 + p1;
#pragma unroll
      for (int off = 1; off < 16; off <<= 1)
        psum += __shfl_xor(psum, off, 32);
      const float alpha = __expf(mrow[r] - mnew);
      lrow[r] = lrow[r] * alpha + psum;
      mrow[r] = mnew;
#pragma unroll
      for (int nt = 0; nt < 8; nt++) O[nt][r] *= alpha;
      Psm[wid][(r + 8 * half) * 32 + lane16] = f2bf(p0);
      Psm[wid][(r + 8 * half) * 32 + 16 + lane16] = f2bf(p1);
    }

    // ---- P*V: batch P + all 8 V-fragment loads, then 8 WMMAs ----
    v8bf plo = *(const v8bf*)&Psm[wid][lane16 * 32 + 8 * half];
    v8bf phi = *(const v8bf*)&Psm[wid][lane16 * 32 + 16 + 8 * half];
    v16bf pf = cat8(plo, phi);
    v16bf vfr[8];
#pragma unroll
    for (int nt = 0; nt < 8; nt++) {
      const __bf16* vrow = &VTs[cur][(nt * 16 + lane16) * LDV];
      v8bf lo = *(const v8bf*)&vrow[8 * half];
      v8bf hi = *(const v8bf*)&vrow[16 + 8 * half];
      vfr[nt] = cat8(lo, hi);
    }
#pragma unroll
    for (int nt = 0; nt < 8; nt++)
      O[nt] = wmma_bf16(pf, vfr[nt], O[nt]);
  }

  // normalize + store ctx[b, s, head*HD + hd]
  float rl[8];
#pragma unroll
  for (int r = 0; r < 8; r++) rl[r] = 1.0f / lrow[r];
#pragma unroll
  for (int nt = 0; nt < 8; nt++) {
#pragma unroll
    for (int r = 0; r < 8; r++) {
      const int srow = qrow0 + r + 8 * half;
      ctx[((size_t)(b * Sc + srow) * NHc + head) * HDc + nt * 16 + lane16] =
          f2bf(O[nt][r] * rl[r]);
    }
  }
}

// ---------------- launch ----------------
extern "C" void kernel_launch(void* const* d_in, const int* in_sizes, int n_in,
                              void* d_out, int out_size, void* d_ws,
                              size_t ws_size, hipStream_t stream) {
  (void)in_sizes; (void)n_in; (void)out_size; (void)ws_size;
  const float* hs      = (const float*)d_in[0];
  const float* w_qkv   = (const float*)d_in[1];
  const float* b_qkv   = (const float*)d_in[2];
  const float* w_dense = (const float*)d_in[3];
  const int*   pos     = (const int*)d_in[4];
  float* out = (float*)d_out;

  char* ws = (char*)d_ws;
  size_t off = 0;
  __bf16* hsb   = (__bf16*)(ws + off); off += (size_t)Mrows * Hc * 2;   // 32 MB
  __bf16* wqkvb = (__bf16*)(ws + off); off += (size_t)Hc * Nqkv * 2;    // 36 MB
  __bf16* qkvb  = (__bf16*)(ws + off); off += (size_t)Mrows * Nqkv * 2; // 36 MB
  __bf16* Qb    = (__bf16*)(ws + off); off += (size_t)Bc * NHc * Sc * HDc * 2;  // 32 MB
  __bf16* Kb    = (__bf16*)(ws + off); off += (size_t)Bc * NKVc * Sc * HDc * 2; // 2 MB
  __bf16* Vb    = (__bf16*)(ws + off); off += (size_t)Bc * NKVc * Sc * HDc * 2; // 2 MB
  __bf16* wdenseb = hsb;   // reuse: hidden bf16 dead after GEMM1
  __bf16* ctx     = wqkvb; // reuse: w_qkv bf16 dead after GEMM1

  cvt_f32_bf16<<<2048, 256, 0, stream>>>(hs, hsb, (size_t)Mrows * Hc);
  cvt_f32_bf16<<<2048, 256, 0, stream>>>(w_qkv, wqkvb, (size_t)Hc * Nqkv);

  gemm_wmma<true, true><<<dim3(Nqkv / 128, Mrows / 128), 256, 0, stream>>>(
      hsb, wqkvb, b_qkv, (void*)qkvb, Mrows, Nqkv, Hc);

  rope_split<<<Mrows, 128, 0, stream>>>(qkvb, pos, Qb, Kb, Vb);

  cvt_f32_bf16<<<2048, 256, 0, stream>>>(w_dense, wdenseb, (size_t)Hc * Hc);

  attn_flash<<<dim3(Sc / 64, NHc, Bc), 128, 0, stream>>>(Qb, Kb, Vb, ctx);

  gemm_wmma<false, false><<<dim3(Hc / 128, Mrows / 128), 256, 0, stream>>>(
      ctx, wdenseb, nullptr, (void*)out, Mrows, Hc, Hc);
}
